// VectorQuantizer2_3539053052640
// MI455X (gfx1250) — compile-verified, matrix operand order checked by reading, not run on hardware
//
#include <hip/hip_runtime.h>
#include <hip/hip_bf16.h>

// ---------------------------------------------------------------------------
// VQ-VAE vector quantizer for MI455X (gfx1250).
//   d(n,e) = ||z_n||^2 + ||e||^2 - 2 z_n.e   -> argmin_e  (||z||^2 dropped)
// GEMM 16384x8192x256 via v_wmma_f32_16x16x32_bf16 (hi/lo bf16 split, 3 WMMAs
// per tile for ~fp32 ranking accuracy).  B tiles + ||e||^2 staged into a
// double-buffered LDS through GLOBAL_LOAD_ASYNC_TO_LDS_* (ASYNCcnt), so the
// next tile's fill overlaps the current tile's 48 WMMAs.
// ---------------------------------------------------------------------------

#define NB    16
#define NC    256
#define NH    32
#define NW    32
#define NROWS 16384      // B*H*W
#define EDIM  256
#define NE    8192

#define ZQ_ELEMS 4194304          // 16*256*32*32
#define LOSS_OFF 4194304
#define IDX_OFF  4194305

typedef __attribute__((ext_vector_type(16))) __bf16 bf16x16;
typedef __attribute__((ext_vector_type(8)))  float  f32x8;

union V16B { uint4 u[2]; bf16x16 v; };

typedef __attribute__((address_space(3))) void lds_void;

__device__ __forceinline__ void async_load_b128(void* lds, const void* gmem) {
  asm volatile("global_load_async_to_lds_b128 %0, %1, off"
               :: "v"((lds_void*)lds), "v"(gmem) : "memory");
}
__device__ __forceinline__ void async_load_b32(void* lds, const void* gmem) {
  asm volatile("global_load_async_to_lds_b32 %0, %1, off"
               :: "v"((lds_void*)lds), "v"(gmem) : "memory");
}
__device__ __forceinline__ void wait_async0() {
  asm volatile("s_wait_asynccnt 0x0" ::: "memory");
}

__device__ __forceinline__ unsigned short f32_to_bf16_rn(float f) {
  unsigned int u = __float_as_uint(f);
  unsigned int r = u + 0x7FFFu + ((u >> 16) & 1u);
  return (unsigned short)(r >> 16);
}
__device__ __forceinline__ float bf16_to_f32(unsigned short h) {
  return __uint_as_float(((unsigned int)h) << 16);
}

// --- pass 1: codebook -> bf16 hi/lo + ||e||^2 -------------------------------
__global__ void __launch_bounds__(256)
vq_prep_cb(const float* __restrict__ cbk,
           unsigned short* __restrict__ cb_hi,
           unsigned short* __restrict__ cb_lo,
           float* __restrict__ e_norm) {
  __shared__ float red[256];
  const int j = blockIdx.x;          // code id
  const int t = threadIdx.x;         // k
  const float v = cbk[(size_t)j * EDIM + t];
  const unsigned short h = f32_to_bf16_rn(v);
  const float hv = bf16_to_f32(h);
  cb_hi[(size_t)j * EDIM + t] = h;
  cb_lo[(size_t)j * EDIM + t] = f32_to_bf16_rn(v - hv);
  red[t] = v * v;
  __syncthreads();
  for (int s = 128; s > 0; s >>= 1) {
    if (t < s) red[t] += red[t + s];
    __syncthreads();
  }
  if (t == 0) e_norm[j] = red[0];
}

// --- pass 2: z (B,C,H,W) -> z_flat (N,256) bf16 hi/lo -----------------------
__global__ void __launch_bounds__(256)
vq_prep_z(const float* __restrict__ z,
          unsigned short* __restrict__ z_hi,
          unsigned short* __restrict__ z_lo) {
  const int n = blockIdx.x;          // flat row, n = b*1024 + h*32 + w
  const int c = threadIdx.x;
  const int b = n >> 10;
  const int hw = n & 1023;
  const float v = z[((size_t)(b * NC + c) << 10) + hw];
  const unsigned short h = f32_to_bf16_rn(v);
  z_hi[(size_t)n * EDIM + c] = h;
  z_lo[(size_t)n * EDIM + c] = f32_to_bf16_rn(v - bf16_to_f32(h));
}

// --- pass 3: WMMA distance GEMM + running argmin ----------------------------
#define TILE_CODES 32
#define LDS_K 264   // padded row (528 B = 132 dwords -> 16 rows hit 16 banks)
#define NITER (NE / TILE_CODES)

__global__ void __launch_bounds__(256)
vq_argmin(const unsigned short* __restrict__ z_hi,
          const unsigned short* __restrict__ z_lo,
          const unsigned short* __restrict__ cb_hi,
          const unsigned short* __restrict__ cb_lo,
          const float* __restrict__ e_norm,
          int* __restrict__ idx_out) {
  __shared__ unsigned short sB[2][2][TILE_CODES][LDS_K]; // [buf][hi|lo][code][k]
  __shared__ float sEn[2][TILE_CODES];                   // [buf][code]

  const int lane = threadIdx.x & 31;
  const int rowbase = blockIdx.x * 128 + (threadIdx.x >> 5) * 16;
  const int lsel = lane & 15;
  const int ksel = (lane >> 4) * 8;   // which 8-elem K half this lane packs

  // A operands (16 rows x 256 K, hi+lo) resident in VGPRs.
  V16B Ahi[8], Alo[8];
  {
    const unsigned short* ph = z_hi + (size_t)(rowbase + lsel) * EDIM;
    const unsigned short* pl = z_lo + (size_t)(rowbase + lsel) * EDIM;
#pragma unroll
    for (int ch = 0; ch < 8; ++ch) {
      const int k0 = ch * 32 + ksel;
      Ahi[ch].u[0] = *(const uint4*)(ph + k0);
      Ahi[ch].u[1] = *(const uint4*)(ph + k0 + 16);
      Alo[ch].u[0] = *(const uint4*)(pl + k0);
      Alo[ch].u[1] = *(const uint4*)(pl + k0 + 16);
    }
  }

  // async stage of one 32-code tile (hi+lo+norms) into LDS buffer `buf`
  auto stage = [&](int cb, int buf) {
    const int t = threadIdx.x;
#pragma unroll
    for (int i = 0; i < 4; ++i) {
      const int e = t + i * 256;      // [0,1024)
      const int row = e >> 5;
      const int k8  = (e & 31) * 8;
      async_load_b128(&sB[buf][0][row][k8],
                      cb_hi + (size_t)(cb + row) * EDIM + k8);
      async_load_b128(&sB[buf][1][row][k8],
                      cb_lo + (size_t)(cb + row) * EDIM + k8);
    }
    if (t < TILE_CODES)
      async_load_b32(&sEn[buf][t], e_norm + cb + t);
  };

  float best[8];
  int   bidx[8];
#pragma unroll
  for (int k = 0; k < 8; ++k) { best[k] = 3.4e38f; bidx[k] = 0x7fffffff; }

  stage(0, 0);   // prologue fill

  for (int it = 0; it < NITER; ++it) {
    const int buf = it & 1;
    wait_async0();        // this wave's tile-`it` async writes have landed
    __syncthreads();      // ...and everyone else's; also guards buf reuse
    if (it + 1 < NITER) stage((it + 1) * TILE_CODES, (it + 1) & 1);

#pragma unroll
    for (int sub = 0; sub < 2; ++sub) {
      const int crow = sub * 16 + lsel;
      f32x8 acc = {0.f, 0.f, 0.f, 0.f, 0.f, 0.f, 0.f, 0.f};
#pragma unroll
      for (int ch = 0; ch < 8; ++ch) {
        V16B Bh, Bl;
        const int k0 = ch * 32 + ksel;
        Bh.u[0] = *(const uint4*)&sB[buf][0][crow][k0];
        Bh.u[1] = *(const uint4*)&sB[buf][0][crow][k0 + 16];
        Bl.u[0] = *(const uint4*)&sB[buf][1][crow][k0];
        Bl.u[1] = *(const uint4*)&sB[buf][1][crow][k0 + 16];
        // dot ~= hi*hi + hi*lo + lo*hi  (lo*lo negligible)
        acc = __builtin_amdgcn_wmma_f32_16x16x32_bf16(false, Ahi[ch].v, false, Bh.v, (short)0, acc, false, false);
        acc = __builtin_amdgcn_wmma_f32_16x16x32_bf16(false, Ahi[ch].v, false, Bl.v, (short)0, acc, false, false);
        acc = __builtin_amdgcn_wmma_f32_16x16x32_bf16(false, Alo[ch].v, false, Bh.v, (short)0, acc, false, false);
      }
      const int code = it * TILE_CODES + sub * 16 + lsel;  // this lane's column
      const float en = sEn[buf][sub * 16 + lsel];
#pragma unroll
      for (int k = 0; k < 8; ++k) {            // rows k (+8 for upper half)
        const float d = fmaf(-2.0f, acc[k], en);
        if (d < best[k]) { best[k] = d; bidx[k] = code; }
      }
    }
  }

  // argmin across the 16 lanes holding the 16 column-classes of each row;
  // tie-break on smaller index to match jnp.argmin (first minimum).
#pragma unroll
  for (int m = 1; m < 16; m <<= 1) {
#pragma unroll
    for (int k = 0; k < 8; ++k) {
      const float ov = __shfl_xor(best[k], m, 32);
      const int   oi = __shfl_xor(bidx[k], m, 32);
      if (ov < best[k] || (ov == best[k] && oi < bidx[k])) { best[k] = ov; bidx[k] = oi; }
    }
  }
  if (lsel == 0) {
    const int r = rowbase + (lane >> 4) * 8;
#pragma unroll
    for (int k = 0; k < 8; ++k) idx_out[r + k] = bidx[k];
  }
}

// --- pass 4: gather z_q, emit idx, per-row loss partials --------------------
__global__ void __launch_bounds__(256)
vq_gather(const float* __restrict__ z,
          const float* __restrict__ cbk,
          const int* __restrict__ idx,
          float* __restrict__ out,
          float* __restrict__ partial) {
  __shared__ float red[256];
  const int n = blockIdx.x;
  const int c = threadIdx.x;
  const int b = n >> 10;
  const int hw = n & 1023;
  const int id = idx[n];
  const float q  = cbk[(size_t)id * EDIM + c];
  const float zv = z[((size_t)(b * NC + c) << 10) + hw];
  out[((size_t)(b * NC + c) << 10) + hw] = q;     // z_q back in (B,C,H,W)
  const float dlt = q - zv;
  red[c] = dlt * dlt;
  __syncthreads();
  for (int s = 128; s > 0; s >>= 1) {
    if (c < s) red[c] += red[c + s];
    __syncthreads();
  }
  if (c == 0) {
    partial[n] = red[0];
    out[IDX_OFF + n] = (float)id;                 // idx as float slot
  }
}

__global__ void __launch_bounds__(256)
vq_loss(const float* __restrict__ partial, float* __restrict__ out) {
  __shared__ float red[256];
  const int t = threadIdx.x;
  float s = 0.f;
  for (int i = t; i < NROWS; i += 256) s += partial[i];   // fixed order
  red[t] = s;
  __syncthreads();
  for (int st = 128; st > 0; st >>= 1) {
    if (t < st) red[t] += red[t + st];
    __syncthreads();
  }
  // loss = mean + BETA*mean = 1.25 * sum / (N*EDIM)
  if (t == 0) out[LOSS_OFF] = 1.25f * red[0] / 4194304.0f;
}

// ---------------------------------------------------------------------------
extern "C" void kernel_launch(void* const* d_in, const int* in_sizes, int n_in,
                              void* d_out, int out_size, void* d_ws, size_t ws_size,
                              hipStream_t stream) {
  const float* z   = (const float*)d_in[0];
  const float* cbk = (const float*)d_in[1];
  float* out = (float*)d_out;

  char* ws = (char*)d_ws;
  float*          e_norm  = (float*)(ws + 0);                    //  32 KB
  int*            idxbuf  = (int*)(ws + 32768);                  //  64 KB
  float*          partial = (float*)(ws + 98304);                //  64 KB
  unsigned short* z_hi    = (unsigned short*)(ws + 163840);      //   8 MB
  unsigned short* z_lo    = (unsigned short*)(ws + 163840 + 8388608);
  unsigned short* cb_hi   = (unsigned short*)(ws + 163840 + 16777216);
  unsigned short* cb_lo   = (unsigned short*)(ws + 163840 + 16777216 + 4194304);

  vq_prep_cb<<<NE, 256, 0, stream>>>(cbk, cb_hi, cb_lo, e_norm);
  vq_prep_z<<<NROWS, 256, 0, stream>>>(z, z_hi, z_lo);
  vq_argmin<<<NROWS / 128, 256, 0, stream>>>(z_hi, z_lo, cb_hi, cb_lo, e_norm, idxbuf);
  vq_gather<<<NROWS, 256, 0, stream>>>(z, cbk, idxbuf, out, partial);
  vq_loss<<<1, 256, 0, stream>>>(partial, out);
}
